// Encoder_8469675508282
// MI455X (gfx1250) — matrix operand order repaired and checked by reading, not compiled, not run on hardware
//
#include <hip/hip_runtime.h>

typedef __attribute__((ext_vector_type(16))) _Float16 hv16;
typedef __attribute__((ext_vector_type(8)))  _Float16 hv8;
typedef __attribute__((ext_vector_type(8)))  float    fv8;

#define SEQ_LEN 512
#define BATCH_N 256
#define HID     256
#define THREE_H 768
#define WN      (THREE_H * HID)   // elements per weight tensor
#define NWG     48
#define APITCH  264               // f16 per LDS A-row (256 + 8 pad -> conflict-free b128 reads)

__device__ __forceinline__ float sigm_f(float v) { return 1.0f / (1.0f + __expf(-v)); }
__device__ __forceinline__ float tanh_f(float v) {
  float a = fminf(fmaxf(v, -15.0f), 15.0f);
  float e = __expf(2.0f * a);
  return (e - 1.0f) / (e + 1.0f);
}

// Device-wide sense-free barrier: monotonic counter, goal = barrier_index * NWG.
__device__ __forceinline__ void grid_barrier(unsigned* cnt, unsigned goal) {
  __threadfence();
  __syncthreads();
  if (threadIdx.x == 0) {
    __hip_atomic_fetch_add(cnt, 1u, __ATOMIC_ACQ_REL, __HIP_MEMORY_SCOPE_AGENT);
    while (__hip_atomic_load(cnt, __ATOMIC_ACQUIRE, __HIP_MEMORY_SCOPE_AGENT) < goal)
      __builtin_amdgcn_s_sleep(1);
  }
  __syncthreads();
}

// One wave: C[16x16]x2 tiles = A(16x256, f16 in LDS) * W^T, f32 accum, + bias.
// A-fragment (contiguous K, matching B): lanes 0-15 hold K{0..15}, lanes 16-31 K{16..31}.
// B-fragment: lane (n = lid&15) holds column n, K chunk selected by lid>>4 -> contiguous 32B of W row.
__device__ __forceinline__ void gemm_pair(const _Float16* aT, const _Float16* __restrict__ W,
                                          const float* __restrict__ bias, float* __restrict__ outb,
                                          int mt, int nt0, int nt1) {
  const int lid = threadIdx.x & 31;
  const int n   = lid & 15;
  const int kh  = lid >> 4;
  fv8 acc0, acc1;
  const float b0 = bias[nt0 * 16 + n];
  const float b1 = bias[nt1 * 16 + n];
#pragma unroll
  for (int r = 0; r < 8; ++r) { acc0[r] = b0; acc1[r] = b1; }
  const _Float16* arow = aT + (size_t)n * APITCH;
  const _Float16* w0 = W + (size_t)(nt0 * 16 + n) * HID + kh * 16;
  const _Float16* w1 = W + (size_t)(nt1 * 16 + n) * HID + kh * 16;
#pragma unroll
  for (int k8 = 0; k8 < 8; ++k8) {
    const int k0 = k8 * 32;
    hv8 lo = *(const hv8*)(arow + k0 + kh * 16);
    hv8 hi = *(const hv8*)(arow + k0 + kh * 16 + 8);
    hv16 a = __builtin_shufflevector(lo, hi, 0,1,2,3,4,5,6,7,8,9,10,11,12,13,14,15);
    hv16 bb0 = *(const hv16*)(w0 + k0);
    hv16 bb1 = *(const hv16*)(w1 + k0);
    acc0 = __builtin_amdgcn_wmma_f32_16x16x32_f16(false, a, false, bb0, (short)0, acc0, false, false);
    acc1 = __builtin_amdgcn_wmma_f32_16x16x32_f16(false, a, false, bb1, (short)0, acc1, false, false);
  }
  const int row0 = mt * 16 + 8 * kh;  // C layout: vgpr r -> row r (+8 for upper half-lanes), col = lane&15
  float* o0 = outb + (size_t)row0 * THREE_H + nt0 * 16 + n;
  float* o1 = outb + (size_t)row0 * THREE_H + nt1 * 16 + n;
#pragma unroll
  for (int r = 0; r < 8; ++r) {
    o0[(size_t)r * THREE_H] = acc0[r];
    o1[(size_t)r * THREE_H] = acc1[r];
  }
}

__global__ void init_ctl_kernel(unsigned* ctl) {
  if (threadIdx.x < 8) ctl[threadIdx.x] = 0u;
}

__global__ void absmax_kernel(const float* __restrict__ w, unsigned* dst, int n) {
  __shared__ float red[256];
  float m = 0.0f;
  for (int i = blockIdx.x * blockDim.x + threadIdx.x; i < n; i += gridDim.x * blockDim.x)
    m = fmaxf(m, fabsf(w[i]));
  red[threadIdx.x] = m;
  __syncthreads();
  for (int s = 128; s > 0; s >>= 1) {
    if ((int)threadIdx.x < s) red[threadIdx.x] = fmaxf(red[threadIdx.x], red[threadIdx.x + s]);
    __syncthreads();
  }
  if (threadIdx.x == 0) atomicMax(dst, __float_as_uint(red[0]));
}

// Symmetric per-tensor int8 fake-quant (round-half-even == jnp.round), emitted as f16.
__global__ void quant_kernel(const float* __restrict__ w, _Float16* __restrict__ wq,
                             const unsigned* maxbits, int n) {
  const float mx = fmaxf(__uint_as_float(*maxbits), 1e-30f);
  const float scale = mx * (1.0f / 127.0f);
  const float inv = 127.0f / mx;
  for (int i = blockIdx.x * blockDim.x + threadIdx.x; i < n; i += gridDim.x * blockDim.x)
    wq[i] = (_Float16)(rintf(w[i] * inv) * scale);
}

__global__ __launch_bounds__(256) void gru2_persistent(
    const float* __restrict__ x,
    const float* __restrict__ h1_init, const float* __restrict__ h2_init,
    const float* __restrict__ b_ih1, const float* __restrict__ b_hh1,
    const float* __restrict__ b_ih2, const float* __restrict__ b_hh2,
    const _Float16* __restrict__ q_ih1, const _Float16* __restrict__ q_hh1,
    const _Float16* __restrict__ q_ih2, const _Float16* __restrict__ q_hh2,
    float* __restrict__ gi1, float* __restrict__ gh1,
    float* __restrict__ gi2, float* __restrict__ gh2,
    float* __restrict__ h1s, float* __restrict__ h2s, _Float16* __restrict__ h1h,
    unsigned* barCnt, float* __restrict__ out)
{
  __shared__ __attribute__((aligned(64))) _Float16 aT[16 * APITCH];
  const int w = blockIdx.x, tid = threadIdx.x;
  const int gX = w >> 4, mtX = w & 15;       // Phase X: 3 gemms x 16 M-tiles
  const int mtY = w / 3, subY = w - mtY * 3; // Phase Y: 16 M-tiles x 3 N-chunks
  const int wv = tid >> 5;
  unsigned barId = 0;

  for (int t = 0; t < SEQ_LEN; ++t) {
    grid_barrier(barCnt, ++barId * NWG);
    // -------- Phase X: gi1 = x_t*Wih1^T+b, gh1 = h1(t-1)*Whh1^T+b, gh2 = h2(t-1)*Whh2^T+b
    if (gX == 0) {
      const float* xt = x + (size_t)t * BATCH_N * HID + (size_t)mtX * 16 * HID;
      for (int i = tid; i < 16 * HID; i += 256) {
        int r = i >> 8, c = i & 255;
        aT[r * APITCH + c] = (_Float16)xt[r * HID + c];
      }
    } else if (gX == 1) {
      for (int i = tid; i < 16 * HID; i += 256) {
        int r = i >> 8, c = i & 255, R = mtX * 16 + r;
        aT[r * APITCH + c] = (t == 0) ? (_Float16)h1_init[R * HID + c] : h1h[R * HID + c];
      }
    } else {
      // Materialize h2(t-1) from last step's gate buffers (unique owner of these rows).
      for (int i = tid; i < 16 * HID; i += 256) {
        int r = i >> 8, c = i & 255, R = mtX * 16 + r;
        float hnew;
        if (t == 0) {
          hnew = h2_init[R * HID + c];
        } else {
          const float* gi = gi2 + (size_t)R * THREE_H;
          const float* gh = gh2 + (size_t)R * THREE_H;
          float rg = sigm_f(gi[c] + gh[c]);
          float zg = sigm_f(gi[HID + c] + gh[HID + c]);
          float ng = tanh_f(gi[2 * HID + c] + rg * gh[2 * HID + c]);
          float hp = h2s[((t + 1) & 1) * BATCH_N * HID + R * HID + c];
          hnew = (1.0f - zg) * ng + zg * hp;
        }
        h2s[(t & 1) * BATCH_N * HID + R * HID + c] = hnew;
        aT[r * APITCH + c] = (_Float16)hnew;
      }
    }
    __syncthreads();
    {
      const _Float16* W = (gX == 0) ? q_ih1 : (gX == 1) ? q_hh1 : q_hh2;
      const float* bias = (gX == 0) ? b_ih1 : (gX == 1) ? b_hh1 : b_hh2;
      float* outb       = (gX == 0) ? gi1  : (gX == 1) ? gh1  : gh2;
#pragma unroll
      for (int p = 0; p < 3; ++p)
        gemm_pair(aT, W, bias, outb, mtX, wv + p * 16, wv + p * 16 + 8);
    }
    grid_barrier(barCnt, ++barId * NWG);
    // -------- Phase Y: h1(t) elementwise (3x redundant per stripe), then gi2 = h1(t)*Wih2^T+b
    for (int i = tid; i < 16 * HID; i += 256) {
      int r = i >> 8, c = i & 255, R = mtY * 16 + r;
      const float* gi = gi1 + (size_t)R * THREE_H;
      const float* gh = gh1 + (size_t)R * THREE_H;
      float rg = sigm_f(gi[c] + gh[c]);
      float zg = sigm_f(gi[HID + c] + gh[HID + c]);
      float ng = tanh_f(gi[2 * HID + c] + rg * gh[2 * HID + c]);
      float hp = (t == 0) ? h1_init[R * HID + c]
                          : h1s[((t + 1) & 1) * BATCH_N * HID + R * HID + c];
      float hnew = (1.0f - zg) * ng + zg * hp;
      if (subY == 0) {
        h1s[(t & 1) * BATCH_N * HID + R * HID + c] = hnew;
        h1h[R * HID + c] = (_Float16)hnew;
      }
      aT[r * APITCH + c] = (_Float16)hnew;
    }
    __syncthreads();
    gemm_pair(aT, q_ih2, b_ih2, gi2, mtY, subY * 16 + wv, subY * 16 + wv + 8);
  }
  grid_barrier(barCnt, ++barId * NWG);
  // -------- Epilogue: h2(511) straight into d_out
  for (int i = w * 256 + tid; i < BATCH_N * HID; i += NWG * 256) {
    int R = i >> 8, c = i & 255;
    const float* gi = gi2 + (size_t)R * THREE_H;
    const float* gh = gh2 + (size_t)R * THREE_H;
    float rg = sigm_f(gi[c] + gh[c]);
    float zg = sigm_f(gi[HID + c] + gh[HID + c]);
    float ng = tanh_f(gi[2 * HID + c] + rg * gh[2 * HID + c]);
    float hp = h2s[1 * BATCH_N * HID + R * HID + c];  // h2(510), written at t=511 (odd buffer)
    out[i] = (1.0f - zg) * ng + zg * hp;
  }
}

extern "C" void kernel_launch(void* const* d_in, const int* in_sizes, int n_in,
                              void* d_out, int out_size, void* d_ws, size_t ws_size,
                              hipStream_t stream) {
  (void)in_sizes; (void)n_in; (void)out_size; (void)ws_size;
  const float* x      = (const float*)d_in[0];
  const float* h1i    = (const float*)d_in[1];
  const float* h2i    = (const float*)d_in[2];
  const float* w_ih1  = (const float*)d_in[3];
  const float* w_hh1  = (const float*)d_in[4];
  const float* b_ih1  = (const float*)d_in[5];
  const float* b_hh1  = (const float*)d_in[6];
  const float* w_ih2  = (const float*)d_in[7];
  const float* w_hh2  = (const float*)d_in[8];
  const float* b_ih2  = (const float*)d_in[9];
  const float* b_hh2  = (const float*)d_in[10];
  float* out = (float*)d_out;

  char* p = (char*)d_ws;
  auto carve = [&](size_t bytes) -> void* {
    void* r = (void*)p;
    p += (bytes + 255) & ~(size_t)255;
    return r;
  };
  _Float16* q_ih1 = (_Float16*)carve(WN * sizeof(_Float16));
  _Float16* q_hh1 = (_Float16*)carve(WN * sizeof(_Float16));
  _Float16* q_ih2 = (_Float16*)carve(WN * sizeof(_Float16));
  _Float16* q_hh2 = (_Float16*)carve(WN * sizeof(_Float16));
  float* gi1 = (float*)carve((size_t)BATCH_N * THREE_H * sizeof(float));
  float* gh1 = (float*)carve((size_t)BATCH_N * THREE_H * sizeof(float));
  float* gi2 = (float*)carve((size_t)BATCH_N * THREE_H * sizeof(float));
  float* gh2 = (float*)carve((size_t)BATCH_N * THREE_H * sizeof(float));
  float* h1s = (float*)carve((size_t)2 * BATCH_N * HID * sizeof(float));
  float* h2s = (float*)carve((size_t)2 * BATCH_N * HID * sizeof(float));
  _Float16* h1h = (_Float16*)carve((size_t)BATCH_N * HID * sizeof(_Float16));
  unsigned* ctl = (unsigned*)carve(256);  // [0..3]=absmax bits, [4]=grid barrier counter

  init_ctl_kernel<<<1, 32, 0, stream>>>(ctl);
  absmax_kernel<<<128, 256, 0, stream>>>(w_ih1, ctl + 0, WN);
  absmax_kernel<<<128, 256, 0, stream>>>(w_hh1, ctl + 1, WN);
  absmax_kernel<<<128, 256, 0, stream>>>(w_ih2, ctl + 2, WN);
  absmax_kernel<<<128, 256, 0, stream>>>(w_hh2, ctl + 3, WN);
  quant_kernel<<<192, 256, 0, stream>>>(w_ih1, q_ih1, ctl + 0, WN);
  quant_kernel<<<192, 256, 0, stream>>>(w_hh1, q_hh1, ctl + 1, WN);
  quant_kernel<<<192, 256, 0, stream>>>(w_ih2, q_ih2, ctl + 2, WN);
  quant_kernel<<<192, 256, 0, stream>>>(w_hh2, q_hh2, ctl + 3, WN);

  gru2_persistent<<<dim3(NWG), dim3(256), 0, stream>>>(
      x, h1i, h2i, b_ih1, b_hh1, b_ih2, b_hh2,
      q_ih1, q_hh1, q_ih2, q_hh2,
      gi1, gh1, gi2, gh2, h1s, h2s, h1h,
      ctl + 4, out);
}
